// ScaledDotProductAttention_26113401159774
// MI455X (gfx1250) — matrix operand order repaired and checked by reading, not compiled, not run on hardware
//
#include <hip/hip_runtime.h>

#define NH    32
#define NKVH  8
#define NREP  4
#define LQ    2048
#define SK    2048
#define DH    128
#define TS    32          // keys per tile
#define LDK   136         // padded bf16 stride for K/V LDS tiles (272B rows)
#define PSTR  72          // padded bf16 stride for P LDS tiles (144B rows)
#define WAVES 8
#define QROWS 16
#define QBLK  (WAVES*QROWS)   // 128 query rows per block

typedef __attribute__((ext_vector_type(16))) __bf16 v16bf;
typedef __attribute__((ext_vector_type(8)))  float  v8f;

union Frag { uint4 u[2]; v16bf v; };

__device__ __forceinline__ unsigned int pack2(float a, float b) {
  union { __bf16 h[2]; unsigned int u; } x;
  x.h[0] = (__bf16)a; x.h[1] = (__bf16)b;
  return x.u;
}

__global__ __launch_bounds__(256, 1)
void ScaledDotProductAttention_26113401159774_kernel(
    const float* __restrict__ Q, const float* __restrict__ K,
    const float* __restrict__ V, const float* __restrict__ Mk,
    float* __restrict__ O)
{
  __shared__ __align__(16) __bf16 ldsK[TS * LDK];
  __shared__ __align__(16) __bf16 ldsV[TS * LDK];
  __shared__ __align__(16) __bf16 ldsP[WAVES * QROWS * PSTR];

  const int tid  = threadIdx.x;
  const int wave = tid >> 5;
  const int lane = tid & 31;
  const int ln   = lane & 15;   // query column in S^T C-layout / A row
  const int hs   = lane >> 4;   // half-wave select

  const int nqb = LQ / QBLK;                  // 16
  const int h   = blockIdx.x / nqb;
  const int qb  = blockIdx.x % nqb;
  const int kvh = h / NREP;
  const int q0  = qb * QBLK + wave * QROWS;

  // fold 1/sqrt(D) and log2(e) so softmax uses exp2 (v_exp_f32 directly)
  const float kSc = 0.08838834764831845f * 1.4426950408889634f;

  const float* Qh = Q  + (size_t)h   * LQ * DH;
  const float* Kh = K  + (size_t)kvh * SK * DH;
  const float* Vh = V  + (size_t)kvh * SK * DH;

  // ---- B-fragments of Q^T, pre-scaled, resident all kernel ----
  // B 32x16: lane%16 = N = query index, element e = K-row = dim 32c + table-order k(e)
  v16bf bq[4];
  #pragma unroll
  for (int c = 0; c < 4; ++c) {
    const float* qp = Qh + (size_t)(q0 + ln) * DH + 32 * c + 8 * hs;
    #pragma unroll
    for (int e = 0; e < 16; ++e)
      bq[c][e] = (__bf16)(qp[e < 8 ? e : 8 + e] * kSc);
  }

  v8f o[8];
  #pragma unroll
  for (int j = 0; j < 8; ++j)
    #pragma unroll
    for (int r = 0; r < 8; ++r) o[j][r] = 0.0f;

  float mcur = -1e30f;
  float lcur = 0.0f;

  __bf16* pl = ldsP + wave * QROWS * PSTR;

  for (int ks = 0; ks < SK; ks += TS) {
    __syncthreads();   // previous iteration done reading ldsK/ldsV

    // ---- cooperative fp32->bf16 staging of K and V tiles ----
    {
      const int row = tid >> 3;
      const int cb  = (tid & 7) * 16;
      const float4* s4 = (const float4*)(Kh + (size_t)(ks + row) * DH + cb);
      float4 a = s4[0], b = s4[1], c = s4[2], d = s4[3];
      unsigned int* w = (unsigned int*)(ldsK + row * LDK + cb);
      w[0]=pack2(a.x,a.y); w[1]=pack2(a.z,a.w); w[2]=pack2(b.x,b.y); w[3]=pack2(b.z,b.w);
      w[4]=pack2(c.x,c.y); w[5]=pack2(c.z,c.w); w[6]=pack2(d.x,d.y); w[7]=pack2(d.z,d.w);
      const float4* t4 = (const float4*)(Vh + (size_t)(ks + row) * DH + cb);
      a = t4[0]; b = t4[1]; c = t4[2]; d = t4[3];
      w = (unsigned int*)(ldsV + row * LDK + cb);
      w[0]=pack2(a.x,a.y); w[1]=pack2(a.z,a.w); w[2]=pack2(b.x,b.y); w[3]=pack2(b.z,b.w);
      w[4]=pack2(c.x,c.y); w[5]=pack2(c.z,c.w); w[6]=pack2(d.x,d.y); w[7]=pack2(d.z,d.w);
    }
    if (ks + TS < SK) {   // warm L2/L0 for the next tile
      __builtin_prefetch(Kh + (size_t)(ks + TS) * DH + tid * 16, 0, 1);
      __builtin_prefetch(Vh + (size_t)(ks + TS) * DH + tid * 16, 0, 1);
    }
    __syncthreads();

    // ---- S^T = K * Q^T, seeded with scaled mask ----
    v8f st[2];
    #pragma unroll
    for (int kt = 0; kt < 2; ++kt) {
      const float* mrow = Mk + (size_t)(q0 + ln) * SK + ks + kt * 16 + 8 * hs;
      float4 m0 = *(const float4*)mrow;
      float4 m1 = *(const float4*)(mrow + 4);
      v8f s;
      s[0]=kSc*m0.x; s[1]=kSc*m0.y; s[2]=kSc*m0.z; s[3]=kSc*m0.w;
      s[4]=kSc*m1.x; s[5]=kSc*m1.y; s[6]=kSc*m1.z; s[7]=kSc*m1.w;
      const __bf16* kb = ldsK + (kt * 16 + ln) * LDK;
      #pragma unroll
      for (int c = 0; c < 4; ++c) {
        Frag f;
        f.u[0] = *(const uint4*)(kb + 32 * c + 8 * hs);
        f.u[1] = *(const uint4*)(kb + 32 * c + 16 + 8 * hs);
        s = __builtin_amdgcn_wmma_f32_16x16x32_bf16(
                false, f.v, false, bq[c], (short)0, s, false, false);
      }
      st[kt] = s;
    }

    // ---- online softmax (query index == ln; halves hold disjoint keys) ----
    float tmax = st[0][0];
    #pragma unroll
    for (int kt = 0; kt < 2; ++kt)
      #pragma unroll
      for (int r = 0; r < 8; ++r) tmax = fmaxf(tmax, st[kt][r]);
    tmax = fmaxf(tmax, __shfl_xor(tmax, 16, 32));
    const float mnew = fmaxf(mcur, tmax);
    const float corr = exp2f(mcur - mnew);

    float rsum = 0.0f;
    #pragma unroll
    for (int kt = 0; kt < 2; ++kt) {
      float p[8];
      #pragma unroll
      for (int r = 0; r < 8; ++r) { p[r] = exp2f(st[kt][r] - mnew); rsum += p[r]; }
      unsigned int* pw = (unsigned int*)(pl + ln * PSTR + kt * 16 + 8 * hs);
      #pragma unroll
      for (int r = 0; r < 4; ++r) pw[r] = pack2(p[2 * r], p[2 * r + 1]);
    }
    rsum += __shfl_xor(rsum, 16, 32);
    lcur = lcur * corr + rsum;
    mcur = mnew;

    // rescale factor per accumulator row: query M = r + 8*hs lives in lane M
    float af[8];
    #pragma unroll
    for (int r = 0; r < 8; ++r) af[r] = __shfl(corr, r + 8 * hs, 32);

    // ---- O += P * V ----
    Frag fa;   // A 16x32: row = query ln, elements over keys
    fa.u[0] = *(const uint4*)(pl + ln * PSTR + 8 * hs);
    fa.u[1] = *(const uint4*)(pl + ln * PSTR + 16 + 8 * hs);
    #pragma unroll
    for (int j = 0; j < 8; ++j) {
      #pragma unroll
      for (int r = 0; r < 8; ++r) o[j][r] *= af[r];
      Frag fb;  // B 32x16: lane%16 = dim 16j+ln, elements = keys in table order
      const __bf16* vb = ldsV + 16 * j + ln;
      #pragma unroll
      for (int s = 0; s < 16; ++s)
        fb.v[s] = vb[(s < 8 ? 8 * hs + s : 8 + 8 * hs + s) * LDK];
      o[j] = __builtin_amdgcn_wmma_f32_16x16x32_bf16(
                 false, fa.v, false, fb.v, (short)0, o[j], false, false);
    }
  }

  // ---- normalize and store ----
  const float inv = 1.0f / lcur;
  float nf[8];
  #pragma unroll
  for (int r = 0; r < 8; ++r) nf[r] = __shfl(inv, r + 8 * hs, 32);
  float* Oh = O + ((size_t)h * LQ + q0) * DH;
  #pragma unroll
  for (int j = 0; j < 8; ++j)
    #pragma unroll
    for (int r = 0; r < 8; ++r)
      Oh[(size_t)(r + 8 * hs) * DH + 16 * j + ln] = o[j][r] * nf[r];
}

extern "C" void kernel_launch(void* const* d_in, const int* in_sizes, int n_in,
                              void* d_out, int out_size, void* d_ws, size_t ws_size,
                              hipStream_t stream) {
  const float* Q  = (const float*)d_in[0];
  const float* K  = (const float*)d_in[1];
  const float* V  = (const float*)d_in[2];
  const float* Mk = (const float*)d_in[3];
  float* O = (float*)d_out;
  dim3 grid(NH * (LQ / QBLK));   // 32 heads * 16 query blocks = 512
  ScaledDotProductAttention_26113401159774_kernel<<<grid, 256, 0, stream>>>(Q, K, V, Mk, O);
}
